// Selector_15333033246761
// MI455X (gfx1250) — matrix operand order repaired and checked by reading, not compiled, not run on hardware
//
#include <hip/hip_runtime.h>

#define N_NODES 100000
#define N_EDGES 1600000
#define IN_CH   128
#define HID     64
#define EDGE_DIM 32
#define LN_EPS  1e-5f

typedef __attribute__((ext_vector_type(16))) _Float16 v16h;
typedef __attribute__((ext_vector_type(8)))  float    v8f;

__device__ __forceinline__ int lane_id() { return (int)(threadIdx.x & 31); }

__device__ __forceinline__ v8f wmma_f16(v16h a, v16h b, v8f c) {
  return __builtin_amdgcn_wmma_f32_16x16x32_f16(false, a, false, b, (short)0, c, false, false);
}

__device__ __forceinline__ void atomic_add_f32(float* p, float v) {
  // relaxed agent scope -> native global_atomic_add_f32 (agg table is L2-resident)
  __hip_atomic_fetch_add(p, v, __ATOMIC_RELAXED, __HIP_MEMORY_SCOPE_AGENT);
}

// ---- CDNA5 async copy: global -> LDS, tracked by ASYNCcnt ----
__device__ __forceinline__ void async_load_b128(unsigned lds_off, const void* gptr) {
  asm volatile("global_load_async_to_lds_b128 %0, %1, off"
               :: "v"(lds_off), "v"(gptr) : "memory");
}
__device__ __forceinline__ void wait_async0() {
  asm volatile("s_wait_asynccnt 0" ::: "memory");
}
__device__ __forceinline__ unsigned lds_off_of(const void* smem) {
  return (unsigned)(unsigned long long)smem;   // generic LDS addr: low 32 bits = LDS offset
}

// =====================================================================
// Packing kernels: f32 row-major -> WMMA-native f16 fragments
// (ISA 7.12.2 layouts). Storage: [frag][lane][16 f16] = 32 B/lane.
// =====================================================================

__global__ __launch_bounds__(256) void pack_A_kernel(
    const float* __restrict__ src, int n_tiles, int ld, int ktiles,
    v16h* __restrict__ out) {
  const int wave = (int)((blockIdx.x * blockDim.x + threadIdx.x) >> 5);
  if (wave >= n_tiles * ktiles) return;
  const int tile = wave / ktiles;
  const int kt   = wave - tile * ktiles;
  const int l = lane_id();
  const int m = l & 15, hi = l >> 4;

  const float* p = src + ((size_t)tile * 16 + m) * ld + kt * 32 + 8 * hi;
  const float4 a0 = *(const float4*)(p);
  const float4 a1 = *(const float4*)(p + 4);
  const float4 a2 = *(const float4*)(p + 16);
  const float4 a3 = *(const float4*)(p + 20);

  v16h v;
  v[0]=(_Float16)a0.x; v[1]=(_Float16)a0.y; v[2]=(_Float16)a0.z; v[3]=(_Float16)a0.w;
  v[4]=(_Float16)a1.x; v[5]=(_Float16)a1.y; v[6]=(_Float16)a1.z; v[7]=(_Float16)a1.w;
  v[8]=(_Float16)a2.x; v[9]=(_Float16)a2.y; v[10]=(_Float16)a2.z; v[11]=(_Float16)a2.w;
  v[12]=(_Float16)a3.x; v[13]=(_Float16)a3.y; v[14]=(_Float16)a3.z; v[15]=(_Float16)a3.w;
  out[(size_t)wave * 32 + l] = v;
}

__global__ __launch_bounds__(256) void pack_B_kernel(
    const float* __restrict__ W, int ktiles, v16h* __restrict__ out) {
  const int wave = (int)((blockIdx.x * blockDim.x + threadIdx.x) >> 5);
  if (wave >= ktiles * 4) return;
  const int kt = wave >> 2, nt = wave & 3;
  const int l = lane_id();
  const int n = l & 15, kk = (l >> 4) * 8;
  v16h b;
#pragma unroll
  for (int e = 0; e < 16; ++e)
    b[e] = (_Float16)W[(kt * 32 + kk + (e & 7) + ((e >> 3) << 4)) * HID + nt * 16 + n];
  out[(size_t)wave * 32 + l] = b;
}

// =====================================================================
// Per-layer node GEMM (templated K-tiles): weights async-staged to LDS;
// xn = h@Wnbr (f16 out) ; hpre = h@Wself + bself ; agg = 0
// =====================================================================
template <int KT>
__global__ __launch_bounds__(256) void node_gemm_kernel(
    const v16h* __restrict__ a16,
    const v16h* __restrict__ wnbr16, const v16h* __restrict__ wself16,
    const float* __restrict__ bself,
    _Float16* __restrict__ xn, float* __restrict__ hpre, float* __restrict__ agg) {
  __shared__ v16h sW[2 * KT * 128];
  const int t = (int)threadIdx.x;
  {
    const unsigned base = lds_off_of(sW);
#pragma unroll
    for (int i = 0; i < KT; ++i) {
      async_load_b128(base + i * 4096 + t * 16, (const char*)wnbr16 + i * 4096 + t * 16);
      async_load_b128(base + (KT + i) * 4096 + t * 16, (const char*)wself16 + i * 4096 + t * 16);
    }
    wait_async0();
    __syncthreads();
  }

  const int wave    = (int)((blockIdx.x * blockDim.x + threadIdx.x) >> 5);
  const int n_tiles = N_NODES / 16;
  if (wave >= n_tiles) return;                 // whole-wave exit (after barrier)
  const int l = lane_id();

  v8f acc_n[4] = {};
  v8f acc_s[4] = {};
#pragma unroll
  for (int kt = 0; kt < KT; ++kt) {
    const v16h a = a16[((size_t)wave * KT + kt) * 32 + l];
#pragma unroll
    for (int nt = 0; nt < 4; ++nt) {
      acc_n[nt] = wmma_f16(a, sW[(kt * 4 + nt) * 32 + l], acc_n[nt]);
      acc_s[nt] = wmma_f16(a, sW[(KT * 4 + kt * 4 + nt) * 32 + l], acc_s[nt]);
    }
  }

  const int row0 = wave * 16;
  const int n = l & 15, hi = l >> 4;
#pragma unroll
  for (int nt = 0; nt < 4; ++nt) {
#pragma unroll
    for (int v = 0; v < 8; ++v) {
      const unsigned idx = (unsigned)(row0 + v + 8 * hi) * HID + (unsigned)(nt * 16 + n);
      xn[idx]   = (_Float16)acc_n[nt][v];
      hpre[idx] = acc_s[nt][v] + bself[nt * 16 + n];
      agg[idx]  = 0.0f;
    }
  }
}

// =====================================================================
// Per-layer edge kernel: msg = relu(xn[src] + edge_attr@Wedge + bmsg);
// agg[dst] += msg. Wedge async-staged to LDS; gathers batched for MLP.
// Grid is exactly 12500 blocks x 8 waves = 100000 tiles: no partial blocks.
// =====================================================================
__global__ __launch_bounds__(256) void edge_kernel(
    const v16h* __restrict__ ea16,
    const int* __restrict__ src, const int* __restrict__ dst,
    const v16h* __restrict__ wedge16, const float* __restrict__ bmsg,
    const _Float16* __restrict__ xn, float* __restrict__ agg) {
  __shared__ v16h sWe[128];
  const int t = (int)threadIdx.x;
  async_load_b128(lds_off_of(sWe) + t * 16, (const char*)wedge16 + t * 16);
  wait_async0();
  __syncthreads();

  const int wave = (int)((blockIdx.x * blockDim.x + threadIdx.x) >> 5);
  const int l = lane_id();

  // K = 32 = EDGE_DIM: one packed A fragment (2 x b128), 4 WMMAs for 64 cols
  const v16h a = ea16[(size_t)wave * 32 + l];
  v8f acc[4] = {};
#pragma unroll
  for (int nt = 0; nt < 4; ++nt)
    acc[nt] = wmma_f16(a, sWe[nt * 32 + l], acc[nt]);

  const int e0 = wave * 16;
  const int n = l & 15, hi = l >> 4;
  unsigned soff[8], doff[8];
#pragma unroll
  for (int v = 0; v < 8; ++v) {
    const int m = v + 8 * hi;
    soff[v] = (unsigned)src[e0 + m] * (unsigned)HID;   // row offsets, unsigned math:
    doff[v] = (unsigned)dst[e0 + m] * (unsigned)HID;   // no sign-extend chains
  }

  // Phase 1: issue all 32 gathers (clustered -> deep memory-level parallelism)
  float xv[4][8];
#pragma unroll
  for (int nt = 0; nt < 4; ++nt) {
    const unsigned col = (unsigned)(nt * 16 + n);
#pragma unroll
    for (int v = 0; v < 8; ++v)
      xv[nt][v] = (float)xn[soff[v] + col];
  }

  // Phase 2: combine + relu + scatter atomics (no-return, STOREcnt-tracked)
#pragma unroll
  for (int nt = 0; nt < 4; ++nt) {
    const unsigned col = (unsigned)(nt * 16 + n);
    const float bm = bmsg[col];
#pragma unroll
    for (int v = 0; v < 8; ++v) {
      float val = acc[nt][v] + xv[nt][v] + bm;
      val = val > 0.0f ? val : 0.0f;
      atomic_add_f32(&agg[doff[v] + col], val);
    }
  }
}

// =====================================================================
// Per-layer combine: h = relu(LN(hpre + agg) * gamma + beta).
// One wave per node; writes h directly in packed A-fragment f16 layout.
// =====================================================================
__device__ __forceinline__ unsigned h16_addr(int node, int c) {
  const int tile = node >> 4, mrow = node & 15;
  const int kt = c >> 5, k = c & 31;
  const int hi = (k >> 3) & 1;
  const int e  = (k & 7) + 8 * (k >> 4);
  return (((unsigned)tile * 2 + kt) * 32 + (mrow + 16 * hi)) * 16 + e;
}

__global__ __launch_bounds__(256) void combine_ln_kernel(
    const float* __restrict__ hpre, const float* __restrict__ agg,
    const float* __restrict__ gamma, const float* __restrict__ beta,
    _Float16* __restrict__ h16out) {
  const int node = (int)((blockIdx.x * blockDim.x + threadIdx.x) >> 5);
  if (node >= N_NODES) return;
  const int l = lane_id();
  const unsigned base = (unsigned)node * HID;

  const float v0 = hpre[base + l]      + agg[base + l];
  const float v1 = hpre[base + 32 + l] + agg[base + 32 + l];

  float s = v0 + v1;
#pragma unroll
  for (int m = 16; m >= 1; m >>= 1) s += __shfl_xor(s, m, 32);
  const float mu = s * (1.0f / HID);

  const float d0 = v0 - mu, d1 = v1 - mu;
  float q = d0 * d0 + d1 * d1;
#pragma unroll
  for (int m = 16; m >= 1; m >>= 1) q += __shfl_xor(q, m, 32);
  const float inv = rsqrtf(q * (1.0f / HID) + LN_EPS);

  float o0 = d0 * inv * gamma[l]      + beta[l];
  float o1 = d1 * inv * gamma[32 + l] + beta[32 + l];
  o0 = o0 > 0.0f ? o0 : 0.0f;
  o1 = o1 > 0.0f ? o1 : 0.0f;
  h16out[h16_addr(node, l)]      = (_Float16)o0;
  h16out[h16_addr(node, 32 + l)] = (_Float16)o1;
}

// =====================================================================
// Head: out = relu(h@W1 + b1) @ W2 + b2. W1 fragments async-staged to LDS.
// =====================================================================
__global__ __launch_bounds__(256) void head_kernel(
    const v16h* __restrict__ h16, const v16h* __restrict__ w116,
    const float* __restrict__ b1, const float* __restrict__ W2,
    const float* __restrict__ b2, float* __restrict__ out) {
  __shared__ v16h sW1[256];
  const int t = (int)threadIdx.x;
  async_load_b128(lds_off_of(sW1) + t * 16, (const char*)w116 + t * 16);
  async_load_b128(lds_off_of(sW1) + 4096 + t * 16, (const char*)w116 + 4096 + t * 16);
  wait_async0();
  __syncthreads();

  const int wave    = (int)((blockIdx.x * blockDim.x + threadIdx.x) >> 5);
  const int n_tiles = N_NODES / 16;
  if (wave >= n_tiles) return;
  const int l = lane_id();

  v8f acc[4] = {};
#pragma unroll
  for (int kt = 0; kt < 2; ++kt) {
    const v16h a = h16[((size_t)wave * 2 + kt) * 32 + l];
#pragma unroll
    for (int nt = 0; nt < 4; ++nt)
      acc[nt] = wmma_f16(a, sW1[(kt * 4 + nt) * 32 + l], acc[nt]);
  }

  const int row0 = wave * 16;
  const int n = l & 15, hi = l >> 4;
  float partial[8];
#pragma unroll
  for (int v = 0; v < 8; ++v) partial[v] = 0.0f;
#pragma unroll
  for (int nt = 0; nt < 4; ++nt) {
    const int col = nt * 16 + n;
    const float w2 = W2[col];
    const float bb = b1[col];
#pragma unroll
    for (int v = 0; v < 8; ++v) {
      float g = acc[nt][v] + bb;
      g = g > 0.0f ? g : 0.0f;
      partial[v] += g * w2;
    }
  }
  const float bias2 = b2[0];
#pragma unroll
  for (int v = 0; v < 8; ++v) {
    float p = partial[v];
    p += __shfl_xor(p, 1, 32);
    p += __shfl_xor(p, 2, 32);
    p += __shfl_xor(p, 4, 32);
    p += __shfl_xor(p, 8, 32);
    if (n == 0) out[row0 + v + 8 * hi] = p + bias2;
  }
}

extern "C" void kernel_launch(void* const* d_in, const int* in_sizes, int n_in,
                              void* d_out, int out_size, void* d_ws, size_t ws_size,
                              hipStream_t stream) {
  (void)in_sizes; (void)n_in; (void)out_size; (void)ws_size;

  // Input order = setup_inputs() dict order (file insertion order)
  const float* x         = (const float*)d_in[0];
  const int*   src       = (const int*)d_in[1];
  const int*   dst       = src + N_EDGES;
  const float* edge_attr = (const float*)d_in[2];

  const float *Wself[3], *Wnbr[3], *Wedge[3], *bmsg[3], *bself[3], *gamma[3], *beta[3];
  for (int i = 0; i < 3; ++i) {
    const int b = 3 + 7 * i;
    Wself[i] = (const float*)d_in[b + 0];
    Wnbr[i]  = (const float*)d_in[b + 1];
    Wedge[i] = (const float*)d_in[b + 2];
    bmsg[i]  = (const float*)d_in[b + 3];
    bself[i] = (const float*)d_in[b + 4];
    gamma[i] = (const float*)d_in[b + 5];
    beta[i]  = (const float*)d_in[b + 6];
  }
  const float* W1 = (const float*)d_in[24];
  const float* b1 = (const float*)d_in[25];
  const float* W2 = (const float*)d_in[26];
  const float* b2 = (const float*)d_in[27];

  // ---- workspace layout (~218 MB) ----
  char* w = (char*)d_ws;
  v16h* ea16 = (v16h*)w;  w += (size_t)(N_EDGES / 16) * 32 * sizeof(v16h);      // 102.4 MB
  v16h* x16  = (v16h*)w;  w += (size_t)(N_NODES / 16) * 4 * 32 * sizeof(v16h);  //  25.6 MB
  v16h* h16A = (v16h*)w;  w += (size_t)(N_NODES / 16) * 2 * 32 * sizeof(v16h);  //  12.8 MB
  v16h* h16B = (v16h*)w;  w += (size_t)(N_NODES / 16) * 2 * 32 * sizeof(v16h);  //  12.8 MB
  _Float16* xn = (_Float16*)w; w += (size_t)N_NODES * HID * sizeof(_Float16);   //  12.8 MB
  float* hpre = (float*)w; w += (size_t)N_NODES * HID * sizeof(float);          //  25.6 MB
  float* agg  = (float*)w; w += (size_t)N_NODES * HID * sizeof(float);          //  25.6 MB
  v16h* wbuf  = (v16h*)w;                                                       //  ~86 KB

  v16h* wn16[3]; v16h* ws16[3]; v16h* we16[3];
  size_t off = 0;
  for (int i = 0; i < 3; ++i) {
    const int kt = (i == 0) ? 4 : 2;
    wn16[i] = wbuf + off; off += (size_t)kt * 4 * 32;
    ws16[i] = wbuf + off; off += (size_t)kt * 4 * 32;
    we16[i] = wbuf + off; off += (size_t)1 * 4 * 32;
  }
  v16h* w116 = wbuf + off;

  const dim3 blk(256);                                 // 8 wave32 per block
  const int node_blocks = ((N_NODES / 16) + 7) / 8;    // 782
  const int edge_blocks = (N_EDGES / 16) / 8;          // 12500 exact
  const int comb_blocks = (N_NODES + 7) / 8;           // 12500

  // ---- one-time packing ----
  pack_A_kernel<<<((N_NODES / 16) * 4 + 7) / 8, blk, 0, stream>>>(x, N_NODES / 16, IN_CH, 4, x16);
  pack_A_kernel<<<((N_EDGES / 16) + 7) / 8, blk, 0, stream>>>(edge_attr, N_EDGES / 16, EDGE_DIM, 1, ea16);
  for (int i = 0; i < 3; ++i) {
    const int kt = (i == 0) ? 4 : 2;
    pack_B_kernel<<<(kt * 4 + 7) / 8, blk, 0, stream>>>(Wnbr[i], kt, wn16[i]);
    pack_B_kernel<<<(kt * 4 + 7) / 8, blk, 0, stream>>>(Wself[i], kt, ws16[i]);
    pack_B_kernel<<<1, blk, 0, stream>>>(Wedge[i], 1, we16[i]);
  }
  pack_B_kernel<<<1, blk, 0, stream>>>(W1, 2, w116);

  // ---- 3 GN layers ----
  const v16h* a16 = x16;
  v16h* hout = h16A;
  for (int layer = 0; layer < 3; ++layer) {
    if (layer == 0)
      node_gemm_kernel<4><<<node_blocks, blk, 0, stream>>>(a16, wn16[layer], ws16[layer],
                                                           bself[layer], xn, hpre, agg);
    else
      node_gemm_kernel<2><<<node_blocks, blk, 0, stream>>>(a16, wn16[layer], ws16[layer],
                                                           bself[layer], xn, hpre, agg);
    edge_kernel<<<edge_blocks, blk, 0, stream>>>(ea16, src, dst, we16[layer],
                                                 bmsg[layer], xn, agg);
    combine_ln_kernel<<<comb_blocks, blk, 0, stream>>>(hpre, agg, gamma[layer], beta[layer],
                                                       (_Float16*)hout);
    a16 = hout;
    hout = (hout == h16A) ? h16B : h16A;
  }

  // ---- head ----
  head_kernel<<<node_blocks, blk, 0, stream>>>(a16, w116, b1, W2, b2, (float*)d_out);
}
